// Aggregate_69973607186597
// MI455X (gfx1250) — matrix operand order repaired and checked by reading, not compiled, not run on hardware
//
#include <hip/hip_runtime.h>

typedef __attribute__((ext_vector_type(16))) __bf16 v16bf;
typedef __attribute__((ext_vector_type(8)))  float  v8f;

#define SLOPE 0.1f

__device__ __forceinline__ __bf16 to_bf16(float x) { return (__bf16)x; }
__device__ __forceinline__ float leaky(float v) { return v >= 0.0f ? v : SLOPE * v; }

struct U8x32 { uint4 a, b; };   // 32 bytes, bit-castable to v16bf

// ---------------------------------------------------------------------------
// DPP butterfly reductions (wave32). After xor1+xor2 the groups of 4 are
// uniform, so row_half_mirror acts as xor4; after that row_mirror acts as
// xor8. ds_swizzle(xor16) crosses the two 16-lane rows.
// ---------------------------------------------------------------------------
template <int CTRL>
__device__ __forceinline__ float dpp_add(float v) {
  int x = __builtin_amdgcn_update_dpp(0, __float_as_int(v), CTRL, 0xF, 0xF, true);
  return v + __int_as_float(x);
}
__device__ __forceinline__ float red16(float v) {   // sum within 16-lane halves
  v = dpp_add<0xB1>(v);    // quad_perm(1,0,3,2)  : xor1
  v = dpp_add<0x4E>(v);    // quad_perm(2,3,0,1)  : xor2
  v = dpp_add<0x141>(v);   // row_half_mirror     : xor4-equivalent
  v = dpp_add<0x140>(v);   // row_mirror          : xor8-equivalent
  return v;
}
__device__ __forceinline__ float red32(float v) {   // full-wave sum
  v = red16(v);
  int x = __builtin_amdgcn_ds_swizzle(__float_as_int(v), 0x401F);  // xor16
  return v + __int_as_float(x);
}

// ---------------------------------------------------------------------------
// Implicit-GEMM conv, NCHW, optional two-input channel concat, bf16 WMMA.
// Block = 128 threads = 4 waves sharing one m-tile; 16x32 weight tile staged
// in LDS per k-chunk. Each wave computes TWO 16x16 n-tiles per A fragment.
// All index math is 32-bit.
// ---------------------------------------------------------------------------
template <int KH, int KW, int PH, int PW>
__global__ void conv2d_wmma(const float* __restrict__ in0, int C0,
                            const float* __restrict__ in1, int C1,
                            const float* __restrict__ Wt, const float* __restrict__ bias,
                            float* __restrict__ out,
                            int Bn, int H, int Wd, int Cout, int relu) {
  constexpr int KHW = KH * KW;
  const int Cin = C0 + C1;
  const int K   = Cin * KHW;
  const int HW  = H * Wd;
  const int N   = Bn * HW;

  __shared__ __align__(16) __bf16 As[16 * 32];

  const int tid   = threadIdx.x;
  const int lane  = tid & 31;
  const int wave  = tid >> 5;
  const int mtile = blockIdx.y;

  // two n-tiles per wave
  const int ncol0 = (blockIdx.x * 8 + wave * 2) * 16 + (lane & 15);
  const int ncol1 = ncol0 + 16;
  const int pc0 = min(ncol0, N - 1);
  const int pc1 = min(ncol1, N - 1);
  const int pb0 = pc0 / HW, py0 = (pc0 % HW) / Wd, px0 = pc0 % Wd;
  const int pb1 = pc1 / HW, py1 = (pc1 % HW) / Wd, px1 = pc1 % Wd;

  // cooperative weight staging: 512 elems, 4 per thread (contiguous k)
  const int sidx = tid * 4;
  const int sm   = sidx >> 5;
  const int sk   = sidx & 31;
  const int mg   = min(mtile * 16 + sm, Cout - 1);
  const float* wsrc = Wt + mg * K;

  const int khalf = (lane >> 4) * 16;
  const int k0B   = (lane >> 4) * 16;
  const int mloc  = lane & 15;

  v8f acc0 = {}, acc1 = {};
  for (int kb = 0; kb < K; kb += 32) {
    __syncthreads();
    {  // stage 16x32 bf16 weight tile into LDS (ds_store_b64)
      float4 w4 = *(const float4*)(wsrc + kb + sk);
      unsigned short h0 = __builtin_bit_cast(unsigned short, to_bf16(w4.x));
      unsigned short h1 = __builtin_bit_cast(unsigned short, to_bf16(w4.y));
      unsigned short h2 = __builtin_bit_cast(unsigned short, to_bf16(w4.z));
      unsigned short h3 = __builtin_bit_cast(unsigned short, to_bf16(w4.w));
      uint2 packed;
      packed.x = (unsigned)h0 | ((unsigned)h1 << 16);
      packed.y = (unsigned)h2 | ((unsigned)h3 << 16);
      *(uint2*)(As + sidx) = packed;
    }
    __syncthreads();

    // A fragment: two ds_load_b128, contiguous k = 16*half + s
    U8x32 u;
    u.a = *(const uint4*)(As + mloc * 32 + khalf);
    u.b = *(const uint4*)(As + mloc * 32 + 8 + khalf);
    v16bf afrag = __builtin_bit_cast(v16bf, u);

    // B fragments: branchless clamped gathers, batched before conversion
    float f0[16], f1[16];
#pragma unroll
    for (int e = 0; e < 16; ++e) {
      int k  = kb + k0B + e;
      int ci = k / KHW;
      int r  = k - ci * KHW;
      int kh = r / KW;
      int kw = r - kh * KW;
      const float* base = (ci < C0) ? in0 : in1;
      int cc   = (ci < C0) ? ci : ci - C0;
      int cper = (ci < C0) ? C0 : C1;
      {
        int yy = py0 + kh - PH, xx = px0 + kw - PW;
        int ok = ((unsigned)yy < (unsigned)H) & ((unsigned)xx < (unsigned)Wd);
        int off = (pb0 * cper + cc) * HW + (ok ? (yy * Wd + xx) : 0);
        float v = base[off];
        f0[e] = ok ? v : 0.0f;
      }
      {
        int yy = py1 + kh - PH, xx = px1 + kw - PW;
        int ok = ((unsigned)yy < (unsigned)H) & ((unsigned)xx < (unsigned)Wd);
        int off = (pb1 * cper + cc) * HW + (ok ? (yy * Wd + xx) : 0);
        float v = base[off];
        f1[e] = ok ? v : 0.0f;
      }
    }
    v16bf bf0, bf1;
#pragma unroll
    for (int e = 0; e < 16; ++e) { bf0[e] = to_bf16(f0[e]); bf1[e] = to_bf16(f1[e]); }

    acc0 = __builtin_amdgcn_wmma_f32_16x16x32_bf16(false, afrag, false, bf0,
                                                   (short)0, acc0, false, false);
    acc1 = __builtin_amdgcn_wmma_f32_16x16x32_bf16(false, afrag, false, bf1,
                                                   (short)0, acc1, false, false);
  }

  // Epilogue: bias hoisted as two float4 loads (clamped so the partial
  // 72-channel m-tile never reads OOB; clamped values are never stored).
  const int mbase  = mtile * 16 + (lane >> 4) * 8;
  const int mclamp = min(mbase, Cout - 8);   // Cout is a multiple of 8 (64/72)
  float4 bv0 = *(const float4*)(bias + mclamp);
  float4 bv1 = *(const float4*)(bias + mclamp + 4);
  float bb[8] = {bv0.x, bv0.y, bv0.z, bv0.w, bv1.x, bv1.y, bv1.z, bv1.w};
  if (ncol0 < N) {
    const int obase = pb0 * Cout * HW + py0 * Wd + px0;
#pragma unroll
    for (int r = 0; r < 8; ++r) {
      int m = mbase + r;
      if (m < Cout) {
        float v = acc0[r] + bb[r];
        if (relu) v = leaky(v);
        out[obase + m * HW] = v;
      }
    }
  }
  if (ncol1 < N) {
    const int obase = pb1 * Cout * HW + py1 * Wd + px1;
#pragma unroll
    for (int r = 0; r < 8; ++r) {
      int m = mbase + r;
      if (m < Cout) {
        float v = acc1[r] + bb[r];
        if (relu) v = leaky(v);
        out[obase + m * HW] = v;
      }
    }
  }
}

// ---------------------------------------------------------------------------
// L2-normalize over channel dim. One thread per pixel.
// ---------------------------------------------------------------------------
__global__ void l2norm_kernel(const float* __restrict__ in, float* __restrict__ out,
                              int Bn, int C, int HW) {
  int t = blockIdx.x * blockDim.x + threadIdx.x;
  if (t >= Bn * HW) return;
  int b = t / HW, s = t % HW;
  float sum = 0.0f;
  for (int c = 0; c < C; ++c) {
    float v = in[(b * C + c) * HW + s];
    sum += v * v;
  }
  float inv = 1.0f / fmaxf(sqrtf(sum), 1e-12f);
  for (int c = 0; c < C; ++c)
    out[(b * C + c) * HW + s] = in[(b * C + c) * HW + s] * inv;
}

// ---------------------------------------------------------------------------
// Correlation (3x3 box over channel dot-products) + online top-2 per pixel.
// One wave per pixel; lanes split 64 channels (2 each). Reference window is
// hoisted; all loads branchless; DPP butterfly for the wave reduction.
// ---------------------------------------------------------------------------
__global__ void corr_topk(const float* __restrict__ refn, const float* __restrict__ nbrn,
                          int* __restrict__ ind,
                          int Bn, int H, int Wd, int p, int ps) {
  const int lane = threadIdx.x & 31;
  const int wave = threadIdx.x >> 5;
  const int pix  = blockIdx.x * 8 + wave;
  const int HW   = H * Wd;
  const int npix = Bn * HW;
  if (pix >= npix) return;
  const int b = pix / HW;
  const int y = (pix % HW) / Wd;
  const int x = pix % Wd;

  const float* rp0 = refn + (b * 64 + lane) * HW;
  const float* rp1 = rp0 + 32 * HW;
  const float* np0 = nbrn + (b * 64 + lane) * HW;
  const float* np1 = np0 + 32 * HW;

  // hoist reference window (invariant over displacements)
  float rv0[9], rv1[9];
#pragma unroll
  for (int w9 = 0; w9 < 9; ++w9) {
    int ki = w9 / 3, kj = w9 % 3;
    int ry = y + ki - 1, rx = x + kj - 1;
    int ok = ((unsigned)ry < (unsigned)H) & ((unsigned)rx < (unsigned)Wd);
    int off = ok ? (ry * Wd + rx) : 0;
    float a = rp0[off], c = rp1[off];
    rv0[w9] = ok ? a : 0.0f;
    rv1[w9] = ok ? c : 0.0f;
  }

  float best1 = -3.0e38f, best2 = -3.0e38f;
  int   d1 = 0, d2 = 0;
  const int pp = p * p;
  for (int d = 0; d < pp; ++d) {
    int row = d / p, col = d % p;
    float s = 0.0f;
#pragma unroll
    for (int w9 = 0; w9 < 9; ++w9) {
      int ki = w9 / 3, kj = w9 % 3;
      int ny = y + ki + row - ps, nx = x + kj + col - ps;
      int ok = ((unsigned)ny < (unsigned)H) & ((unsigned)nx < (unsigned)Wd);
      int off = ok ? (ny * Wd + nx) : 0;
      float a = np0[off], c = np1[off];
      s += rv0[w9] * (ok ? a : 0.0f) + rv1[w9] * (ok ? c : 0.0f);
    }
    s = red32(s);
    if (s > best1)      { best2 = best1; d2 = d1; best1 = s; d1 = d; }
    else if (s > best2) { best2 = s; d2 = d; }
  }
  if (lane == 0) { ind[pix * 2] = d1; ind[pix * 2 + 1] = d2; }
}

// ---------------------------------------------------------------------------
// Per-pixel aggregation GEMM: agg[64 x 9] = W[64 x 1152] x patches[1152 x 9],
// leaky, mask-weighted sum over the 9 positions, leaky, store.
// Block = 256 threads = 8 waves = 2 pixels x 4 co-tiles; 64x32 weight tile
// staged in LDS per k-chunk (shared by all 8 waves).
// ---------------------------------------------------------------------------
__global__ void aggregate_wmma(const float* __restrict__ nbr,
                               const int* __restrict__ ind,
                               const float* __restrict__ mask,   // (B, 72, H, W)
                               const float* __restrict__ Wt,     // (64, 128, 3, 3)
                               const float* __restrict__ bias,   // (64,)
                               float* __restrict__ out,
                               int Bn, int H, int Wd, int p, int ps) {
  const int K = 1152;
  __shared__ __align__(16) __bf16 Ws[64 * 32];   // 4 KB

  const int tid    = threadIdx.x;
  const int lane   = tid & 31;
  const int wave   = tid >> 5;
  const int cotile = wave & 3;
  const int HW     = H * Wd;
  const int npix   = Bn * HW;
  const int pixr   = blockIdx.x * 2 + (wave >> 2);
  const int pvalid = pixr < npix;
  const int pix    = pvalid ? pixr : npix - 1;
  const int b = pix / HW;
  const int y = (pix % HW) / Wd;
  const int x = pix % Wd;

  const int dd0 = ind[pix * 2], dd1 = ind[pix * 2 + 1];
  const int row0 = dd0 / p, col0 = dd0 % p;
  const int row1 = dd1 / p, col1 = dd1 % p;

  // cooperative weight staging: 2048 elems, 8 per thread (contiguous k)
  const int sidx = tid * 8;
  const int sm   = sidx >> 5;
  const int sk   = sidx & 31;
  const float* wsrc = Wt + sm * K;

  const int mA    = cotile * 16 + (lane & 15);
  const int khalf = (lane >> 4) * 16;
  const int k0B   = (lane >> 4) * 16;
  const int pos   = lane & 15;            // conv output position, valid < 9
  const int i = pos / 3, j = pos % 3;
  const int nb0 = b * 64 * HW;

  v8f acc = {};
  for (int kb = 0; kb < K; kb += 32) {
    __syncthreads();
    {  // stage 64x32 bf16 weight tile (ds_store_b128)
      float4 wa = *(const float4*)(wsrc + kb + sk);
      float4 wb = *(const float4*)(wsrc + kb + sk + 4);
      unsigned short h0 = __builtin_bit_cast(unsigned short, to_bf16(wa.x));
      unsigned short h1 = __builtin_bit_cast(unsigned short, to_bf16(wa.y));
      unsigned short h2 = __builtin_bit_cast(unsigned short, to_bf16(wa.z));
      unsigned short h3 = __builtin_bit_cast(unsigned short, to_bf16(wa.w));
      unsigned short h4 = __builtin_bit_cast(unsigned short, to_bf16(wb.x));
      unsigned short h5 = __builtin_bit_cast(unsigned short, to_bf16(wb.y));
      unsigned short h6 = __builtin_bit_cast(unsigned short, to_bf16(wb.z));
      unsigned short h7 = __builtin_bit_cast(unsigned short, to_bf16(wb.w));
      uint4 pk;
      pk.x = (unsigned)h0 | ((unsigned)h1 << 16);
      pk.y = (unsigned)h2 | ((unsigned)h3 << 16);
      pk.z = (unsigned)h4 | ((unsigned)h5 << 16);
      pk.w = (unsigned)h6 | ((unsigned)h7 << 16);
      *(uint4*)(Ws + sidx) = pk;
    }
    __syncthreads();

    // A fragment: two ds_load_b128
    U8x32 u;
    u.a = *(const uint4*)(Ws + mA * 32 + khalf);
    u.b = *(const uint4*)(Ws + mA * 32 + 8 + khalf);
    v16bf afrag = __builtin_bit_cast(v16bf, u);

    // B fragment: branchless clamped gathers, batched before conversion
    float fv[16];
#pragma unroll
    for (int e = 0; e < 16; ++e) {
      int k  = kb + k0B + e;
      int ci = k / 9;
      int r  = k - ci * 9;
      int ki = r / 3, kj = r - ki * 3;
      int t  = ci >> 6;
      int c  = ci & 63;
      int rowt = t ? row1 : row0;
      int colt = t ? col1 : col0;
      int aa = i + ki - 1, bb2 = j + kj - 1;   // patch-local coord (conv pad 1)
      int yy = y + rowt + aa - ps;
      int xx = x + colt + bb2 - ps;
      int ok = (pos < 9) & ((unsigned)aa < 3u) & ((unsigned)bb2 < 3u) &
               ((unsigned)yy < (unsigned)H) & ((unsigned)xx < (unsigned)Wd);
      int off = nb0 + c * HW + (ok ? (yy * Wd + xx) : 0);
      float v = nbr[off];
      fv[e] = ok ? v : 0.0f;
    }
    v16bf bfrag;
#pragma unroll
    for (int e = 0; e < 16; ++e) bfrag[e] = to_bf16(fv[e]);

    acc = __builtin_amdgcn_wmma_f32_16x16x32_bf16(false, afrag, false, bfrag,
                                                  (short)0, acc, false, false);
  }

  // Epilogue: g = m>>3 is constant across the 8 rows of this lane, so the
  // mask value is a single hoisted load; bias is two hoisted float4 loads.
  const int half   = lane >> 4;
  const int mstart = cotile * 16 + 8 * half;
  const int g      = mstart >> 3;
  const int posc   = min(pos, 8);
  float mraw = mask[(b * 72 + g * 9 + posc) * HW + y * Wd + x];
  const float mv = (pos < 9) ? mraw : 0.0f;
  float4 bv0 = *(const float4*)(bias + mstart);
  float4 bv1 = *(const float4*)(bias + mstart + 4);
  float bb[8] = {bv0.x, bv0.y, bv0.z, bv0.w, bv1.x, bv1.y, bv1.z, bv1.w};
  const int obase = b * 64 * HW + y * Wd + x;
#pragma unroll
  for (int r = 0; r < 8; ++r) {
    float v = leaky(acc[r] + bb[r]) * mv;
    v = red16(v);
    if (pvalid && pos == 0)
      out[obase + (mstart + r) * HW] = leaky(v);
  }
}

// ---------------------------------------------------------------------------
// 2x bilinear upsample (half-pixel centers, edge clamp).
// ---------------------------------------------------------------------------
__global__ void up2_bilinear(const float* __restrict__ in, float* __restrict__ out,
                             int Bn, int C, int Hin, int Win) {
  const int Hout = 2 * Hin, Wout = 2 * Win;
  int total = Bn * C * Hout * Wout;
  int t = blockIdx.x * blockDim.x + threadIdx.x;
  if (t >= total) return;
  int ox = t % Wout;
  int oy = (t / Wout) % Hout;
  int bc = t / (Wout * Hout);
  float sy = oy * 0.5f - 0.25f;
  float sx = ox * 0.5f - 0.25f;
  int y0 = (int)floorf(sy), x0 = (int)floorf(sx);
  float wy = sy - y0, wx = sx - x0;
  int y0c = min(max(y0, 0), Hin - 1), y1c = min(max(y0 + 1, 0), Hin - 1);
  int x0c = min(max(x0, 0), Win - 1), x1c = min(max(x0 + 1, 0), Win - 1);
  const float* p = in + bc * Hin * Win;
  float v00 = p[y0c * Win + x0c], v01 = p[y0c * Win + x1c];
  float v10 = p[y1c * Win + x0c], v11 = p[y1c * Win + x1c];
  out[t] = (1.0f - wy) * ((1.0f - wx) * v00 + wx * v01) +
           wy          * ((1.0f - wx) * v10 + wx * v11);
}

// ---------------------------------------------------------------------------
// Host orchestration
// ---------------------------------------------------------------------------
static void launch_conv(hipStream_t s, const float* in0, int C0, const float* in1, int C1,
                        const float* w, const float* b, float* out,
                        int Bn, int H, int W, int Cout, int KH, int KW, int relu) {
  int N = Bn * H * W;
  int ntiles = (N + 15) / 16;
  int mtiles = (Cout + 15) / 16;
  dim3 grid((ntiles + 7) / 8, mtiles, 1);
  if (KH == 3)
    conv2d_wmma<3, 3, 1, 1><<<grid, 128, 0, s>>>(in0, C0, in1, C1, w, b, out, Bn, H, W, Cout, relu);
  else if (KW == 1)
    conv2d_wmma<7, 1, 3, 0><<<grid, 128, 0, s>>>(in0, C0, in1, C1, w, b, out, Bn, H, W, Cout, relu);
  else
    conv2d_wmma<1, 7, 0, 3><<<grid, 128, 0, s>>>(in0, C0, in1, C1, w, b, out, Bn, H, W, Cout, relu);
}

static void launch_l2norm(hipStream_t s, const float* in, float* out, int Bn, int H, int W) {
  int P = Bn * H * W;
  l2norm_kernel<<<(P + 255) / 256, 256, 0, s>>>(in, out, Bn, 64, H * W);
}

static void launch_corr(hipStream_t s, const float* refn, const float* nbrn, int* ind,
                        int Bn, int H, int W, int p) {
  int P = Bn * H * W;
  int ps = 1 + (p - 1) / 2;
  corr_topk<<<(P + 7) / 8, 256, 0, s>>>(refn, nbrn, ind, Bn, H, W, p, ps);
}

static void launch_agg(hipStream_t s, const float* nbr, const int* ind, const float* mask,
                       const float* w, const float* b, float* out,
                       int Bn, int H, int W, int p) {
  int P = Bn * H * W;
  int ps = 1 + (p - 1) / 2;
  aggregate_wmma<<<(P + 1) / 2, 256, 0, s>>>(nbr, ind, mask, w, b, out, Bn, H, W, p, ps);
}

static void launch_up2(hipStream_t s, const float* in, float* out, int Bn, int Hin, int Win) {
  int total = Bn * 64 * 4 * Hin * Win;
  up2_bilinear<<<(total + 255) / 256, 256, 0, s>>>(in, out, Bn, 64, Hin, Win);
}

extern "C" void kernel_launch(void* const* d_in, const int* in_sizes, int n_in,
                              void* d_out, int out_size, void* d_ws, size_t ws_size,
                              hipStream_t stream) {
  (void)in_sizes; (void)n_in; (void)out_size; (void)ws_size;
  // params flattened in setup_inputs() dict order: each entry is (w, b)
  const float* P[56];
  for (int i = 0; i < 56; ++i) P[i] = (const float*)d_in[i];
  const float* L3c1w = P[0];  const float* L3c1b = P[1];
  const float* L3c2w = P[2];  const float* L3c2b = P[3];
  const float* L3c3w = P[4];  const float* L3c3b = P[5];   // 7x1
  const float* L3c4w = P[6];  const float* L3c4b = P[7];   // 1x7
  const float* L3mw  = P[8];  const float* L3mb  = P[9];
  const float* L3nw  = P[10]; const float* L3nb  = P[11];
  const float* L2c1w = P[12]; const float* L2c1b = P[13];
  const float* L2c2w = P[14]; const float* L2c2b = P[15];
  const float* L2c3w = P[16]; const float* L2c3b = P[17];
  const float* L2c4w = P[18]; const float* L2c4b = P[19];  // 7x1
  const float* L2c5w = P[20]; const float* L2c5b = P[21];  // 1x7
  const float* L2mw  = P[22]; const float* L2mb  = P[23];
  const float* L2nw  = P[24]; const float* L2nb  = P[25];
  const float* L2fw  = P[26]; const float* L2fb  = P[27];
  const float* L1c1w = P[28]; const float* L1c1b = P[29];
  const float* L1c2w = P[30]; const float* L1c2b = P[31];
  const float* L1c3w = P[32]; const float* L1c3b = P[33];
  const float* L1c4w = P[34]; const float* L1c4b = P[35];  // 7x1
  const float* L1c5w = P[36]; const float* L1c5b = P[37];  // 1x7
  const float* L1mw  = P[38]; const float* L1mb  = P[39];
  const float* L1nw  = P[40]; const float* L1nb  = P[41];
  const float* L1fw  = P[42]; const float* L1fb  = P[43];
  const float* Cc1w  = P[44]; const float* Cc1b  = P[45];
  const float* Cc2w  = P[46]; const float* Cc2b  = P[47];
  const float* Cc3w  = P[48]; const float* Cc3b  = P[49];  // 7x1
  const float* Cc4w  = P[50]; const float* Cc4b  = P[51];  // 1x7
  const float* Cmw   = P[52]; const float* Cmb   = P[53];
  const float* Cnw   = P[54]; const float* Cnb   = P[55];
  const float* nbr0 = (const float*)d_in[56];
  const float* nbr1 = (const float*)d_in[57];
  const float* nbr2 = (const float*)d_in[58];
  const float* ref0 = (const float*)d_in[59];
  const float* ref1 = (const float*)d_in[60];
  const float* ref2 = (const float*)d_in[61];
  float* outp = (float*)d_out;

  const int B = 2;
  const int H3 = 24, W3 = 24, H2 = 48, W2 = 48, H1 = 96, W1 = 96;
  const size_t S3 = (size_t)B * 64 * H3 * W3;
  const size_t S2 = (size_t)B * 64 * H2 * W2;
  const size_t S1 = (size_t)B * 64 * H1 * W1;
  const size_t M3 = (size_t)B * 72 * H3 * W3;
  const size_t M2 = (size_t)B * 72 * H2 * W2;
  const size_t M1 = (size_t)B * 72 * H1 * W1;

  float* ws = (float*)d_ws;
  size_t o = 0;
  float* t3a    = ws + o; o += S3;
  float* t3b    = ws + o; o += S3;
  float* L3feat = ws + o; o += S3;
  float* mask3  = ws + o; o += M3;
  float* t2a    = ws + o; o += S2;
  float* t2b    = ws + o; o += S2;
  float* L3wup  = ws + o; o += S2;
  float* L3fup  = ws + o; o += S2;
  float* L2feat = ws + o; o += S2;
  float* mask2  = ws + o; o += M2;
  float* t1a    = ws + o; o += S1;
  float* t1b    = ws + o; o += S1;
  float* L2wup  = ws + o; o += S1;
  float* L2fup  = ws + o; o += S1;
  float* L1feat = ws + o; o += S1;
  float* mask1  = ws + o; o += M1;
  int* ind3 = (int*)(ws + o); o += (size_t)B * H3 * W3 * 2;
  int* ind2 = (int*)(ws + o); o += (size_t)B * H2 * W2 * 2;
  int* ind1 = (int*)(ws + o); o += (size_t)B * H1 * W1 * 2;

  // ----- L3 -----
  launch_l2norm(stream, ref2, t3a, B, H3, W3);
  launch_l2norm(stream, nbr2, t3b, B, H3, W3);
  launch_corr(stream, t3a, t3b, ind3, B, H3, W3, 7);
  launch_conv(stream, nbr2, 64, ref2, 64, L3c1w, L3c1b, t3a, B, H3, W3, 64, 3, 3, 1);
  launch_conv(stream, t3a, 64, nullptr, 0, L3c2w, L3c2b, t3b, B, H3, W3, 64, 3, 3, 1);
  launch_conv(stream, t3b, 64, nullptr, 0, L3c3w, L3c3b, t3a, B, H3, W3, 64, 7, 1, 1);
  launch_conv(stream, t3a, 64, nullptr, 0, L3c4w, L3c4b, t3b, B, H3, W3, 64, 1, 7, 1); // L3_w
  launch_conv(stream, t3b, 64, nullptr, 0, L3mw, L3mb, mask3, B, H3, W3, 72, 3, 3, 0);
  launch_agg(stream, nbr2, ind3, mask3, L3nw, L3nb, L3feat, B, H3, W3, 7);
  launch_up2(stream, t3b, L3wup, B, H3, W3);
  launch_up2(stream, L3feat, L3fup, B, H3, W3);

  // ----- L2 -----
  launch_l2norm(stream, ref1, t2a, B, H2, W2);
  launch_l2norm(stream, nbr1, t2b, B, H2, W2);
  launch_corr(stream, t2a, t2b, ind2, B, H2, W2, 11);
  launch_conv(stream, nbr1, 64, ref1, 64, L2c1w, L2c1b, t2a, B, H2, W2, 64, 3, 3, 1);
  launch_conv(stream, t2a, 64, L3wup, 64, L2c2w, L2c2b, t2b, B, H2, W2, 64, 3, 3, 1);
  launch_conv(stream, t2b, 64, nullptr, 0, L2c3w, L2c3b, t2a, B, H2, W2, 64, 3, 3, 1);
  launch_conv(stream, t2a, 64, nullptr, 0, L2c4w, L2c4b, t2b, B, H2, W2, 64, 7, 1, 1);
  launch_conv(stream, t2b, 64, nullptr, 0, L2c5w, L2c5b, t2a, B, H2, W2, 64, 1, 7, 1); // L2_w
  launch_conv(stream, t2a, 64, nullptr, 0, L2mw, L2mb, mask2, B, H2, W2, 72, 3, 3, 0);
  launch_agg(stream, nbr1, ind2, mask2, L2nw, L2nb, t2b, B, H2, W2, 11);
  launch_conv(stream, t2b, 64, L3fup, 64, L2fw, L2fb, L2feat, B, H2, W2, 64, 3, 3, 1);
  launch_up2(stream, t2a, L2wup, B, H2, W2);
  launch_up2(stream, L2feat, L2fup, B, H2, W2);

  // ----- L1 -----
  launch_l2norm(stream, ref0, t1a, B, H1, W1);
  launch_l2norm(stream, nbr0, t1b, B, H1, W1);
  launch_corr(stream, t1a, t1b, ind1, B, H1, W1, 15);
  launch_conv(stream, nbr0, 64, ref0, 64, L1c1w, L1c1b, t1a, B, H1, W1, 64, 3, 3, 1);
  launch_conv(stream, t1a, 64, L2wup, 64, L1c2w, L1c2b, t1b, B, H1, W1, 64, 3, 3, 1);
  launch_conv(stream, t1b, 64, nullptr, 0, L1c3w, L1c3b, t1a, B, H1, W1, 64, 3, 3, 1);
  launch_conv(stream, t1a, 64, nullptr, 0, L1c4w, L1c4b, t1b, B, H1, W1, 64, 7, 1, 1);
  launch_conv(stream, t1b, 64, nullptr, 0, L1c5w, L1c5b, t1a, B, H1, W1, 64, 1, 7, 1); // L1_w
  launch_conv(stream, t1a, 64, nullptr, 0, L1mw, L1mb, mask1, B, H1, W1, 72, 3, 3, 0);
  launch_agg(stream, nbr0, ind1, mask1, L1nw, L1nb, t1b, B, H1, W1, 15);
  launch_conv(stream, t1b, 64, L2fup, 64, L1fw, L1fb, L1feat, B, H1, W1, 64, 3, 3, 1);

  // ----- cascade -----
  launch_conv(stream, L1feat, 64, ref0, 64, Cc1w, Cc1b, t1a, B, H1, W1, 64, 3, 3, 1);
  launch_conv(stream, t1a, 64, nullptr, 0, Cc2w, Cc2b, t1b, B, H1, W1, 64, 3, 3, 1);
  launch_conv(stream, t1b, 64, nullptr, 0, Cc3w, Cc3b, t1a, B, H1, W1, 64, 7, 1, 1);
  launch_conv(stream, t1a, 64, nullptr, 0, Cc4w, Cc4b, t1b, B, H1, W1, 64, 1, 7, 1);
  launch_conv(stream, t1b, 64, nullptr, 0, Cmw, Cmb, mask1, B, H1, W1, 72, 3, 3, 0);
  launch_l2norm(stream, ref0, t1a, B, H1, W1);
  launch_l2norm(stream, L1feat, L2fup, B, H1, W1);
  launch_corr(stream, t1a, L2fup, ind1, B, H1, W1, 15);
  launch_agg(stream, L1feat, ind1, mask1, Cnw, Cnb, outp, B, H1, W1, 15);
}
